// PoseTemporalTransformerEncoder_57140244906177
// MI455X (gfx1250) — compile-verified, matrix operand order checked by reading, not run on hardware
//
#include <hip/hip_runtime.h>
#include <hip/hip_bf16.h>

// ---------------- problem constants ----------------
constexpr int Bc = 16, Tc = 512, Ec = 1024, Hc = 16, FFc = 4096, Lc = 4;
constexpr int HDc = Ec / Hc;                 // 64
constexpr float SCALEc = 0.125f;             // 64^-0.5
constexpr float EPSc = 1e-5f;
constexpr float MAXOFFc = 0.5f;

typedef unsigned short u16;
typedef unsigned int   u32;
typedef unsigned long long u64;
typedef __bf16 bf16x16 __attribute__((ext_vector_type(16)));
typedef float  f32x8   __attribute__((ext_vector_type(8)));

// CDNA5 async global->LDS copy path via inline asm (ISA §15.18.3):
//   GLOBAL_LOAD_ASYNC_TO_LDS_B128: VDST = LDS byte address, VADDR = 64b addr
// tracked by ASYNCcnt, drained with s_wait_asynccnt.
#if defined(__gfx1250__)
#define HAVE_ASYNC_LDS 1
__device__ __forceinline__ void async_g2l_b128(const void* g, void* l) {
  u64 ga = (u64)g;
  u32 la = (u32)(u64)l;                 // flat->LDS: low 32 bits
  asm volatile("global_load_async_to_lds_b128 %0, %1, off"
               :: "v"(la), "v"(ga) : "memory");
}
__device__ __forceinline__ void wait_async0() {
  asm volatile("s_wait_asynccnt 0x0" ::: "memory");
}
#else
#define HAVE_ASYNC_LDS 0
#endif

__device__ __forceinline__ u16 f2bf(float f) {
  u32 u = __float_as_uint(f);
  u32 r = (u + 0x7FFFu + ((u >> 16) & 1u)) >> 16;   // round-to-nearest-even
  return (u16)r;
}
__device__ __forceinline__ u32 pack2(float a, float b) {
  return (u32)f2bf(a) | ((u32)f2bf(b) << 16);
}

// ============================================================
// WMMA GEMM: C = A(bf16|f32) x B(bf16, [N][K] K-contiguous) + epilogue
//   MODE 1: +bias, store bf16 at (b,h,t,hd)     (Q/K projections)
//   MODE 2: *scale + temporal bias, f32 scores  [z][T][T]
//   MODE 3: store bf16 at (b,t,h*HD+hd)         (attn @ V output)
//   MODE 4: +bias +residual, f32 [m][N]         (O-proj / FFN2)
//   MODE 5: +bias, exact GELU, bf16             (FFN1)
//   MODE 6: +bias, store bf16 at (b,h,hd,t)     (V projection, vectorized)
// Double-buffered LDS pipeline; async-to-LDS on gfx1250.
// ============================================================
template<int BM, int BN, int WM, int WN, int MODE, bool A_F32>
__global__ __launch_bounds__(256)
void gemm_wmma(const void* __restrict__ Ag, const u16* __restrict__ Bg,
               const float* __restrict__ bias, const float* __restrict__ extra,
               void* __restrict__ Cg, int K, int lda, int ldb, int ldc,
               long long aSlice, long long bSlice, float scale)
{
  static_assert(WM * WN == 8, "8 waves per block");
  constexpr int BK = 32, LDT = BK + 16;          // 96B row stride (16B aligned)
  constexpr int MI = BM / WM / 16, NI = BN / WN / 16;
  constexpr int AIT = BM * BK / 2048, BIT = BN * BK / 2048;

  __shared__ u16 sA[2][BM * LDT];
  __shared__ u16 sB[2][BN * LDT];

  const int tid = threadIdx.x;
  const int z   = blockIdx.z;
  const int m0  = blockIdx.y * BM;
  const int n0  = blockIdx.x * BN;

  const u16*   A16 = nullptr;
  const float* A32 = nullptr;
  if constexpr (A_F32) A32 = (const float*)Ag + (size_t)z * aSlice;
  else                 A16 = (const u16*)Ag + (size_t)z * aSlice;
  const u16* Bz = Bg + (size_t)z * bSlice;

  const int wave = tid >> 5, lane = tid & 31;
  const int wm = wave % WM, wn = wave / WM;
  const int mW = wm * (BM / WM), nW = wn * (BN / WN);
  const int lrow = lane & 15, half = lane >> 4;

  f32x8 acc[MI][NI] = {};
  uint4 ra[AIT];
#if !HAVE_ASYNC_LDS
  uint4 rb[BIT];
#endif

  // issue loads for K-tile kk0 into buffer nb (regs or async-to-LDS)
  auto stage_issue = [&](int kk0, int nb) {
    if constexpr (A_F32) {
#pragma unroll
      for (int it = 0; it < AIT; ++it) {
        int e = tid * 8 + it * 2048, i = e >> 5, kx = e & 31;
        const float* s = A32 + (size_t)(m0 + i) * lda + (kk0 + kx);
        float4 f0 = *(const float4*)s;
        float4 f1 = *(const float4*)(s + 4);
        ra[it] = make_uint4(pack2(f0.x, f0.y), pack2(f0.z, f0.w),
                            pack2(f1.x, f1.y), pack2(f1.z, f1.w));
      }
    } else {
#if HAVE_ASYNC_LDS
#pragma unroll
      for (int it = 0; it < AIT; ++it) {
        int e = tid * 8 + it * 2048, i = e >> 5, kx = e & 31;
        async_g2l_b128(A16 + (size_t)(m0 + i) * lda + (kk0 + kx),
                       &sA[nb][i * LDT + kx]);
      }
#else
#pragma unroll
      for (int it = 0; it < AIT; ++it) {
        int e = tid * 8 + it * 2048, i = e >> 5, kx = e & 31;
        ra[it] = *(const uint4*)(A16 + (size_t)(m0 + i) * lda + (kk0 + kx));
      }
      __builtin_prefetch(A16 + (size_t)(m0 + (tid & (BM - 1))) * lda + kk0 + BK, 0, 1);
#endif
    }
#if HAVE_ASYNC_LDS
#pragma unroll
    for (int it = 0; it < BIT; ++it) {
      int e = tid * 8 + it * 2048, j = e >> 5, kx = e & 31;
      async_g2l_b128(Bz + (size_t)(n0 + j) * ldb + (kk0 + kx),
                     &sB[nb][j * LDT + kx]);
    }
#else
#pragma unroll
    for (int it = 0; it < BIT; ++it) {
      int e = tid * 8 + it * 2048, j = e >> 5, kx = e & 31;
      rb[it] = *(const uint4*)(Bz + (size_t)(n0 + j) * ldb + (kk0 + kx));
    }
#endif
  };

  // make buffer nb consistent (write regs / drain async)
  auto stage_commit = [&](int nb) {
    if constexpr (A_F32) {
#pragma unroll
      for (int it = 0; it < AIT; ++it) {
        int e = tid * 8 + it * 2048, i = e >> 5, kx = e & 31;
        *(uint4*)&sA[nb][i * LDT + kx] = ra[it];
      }
    }
#if HAVE_ASYNC_LDS
    wait_async0();
#else
    if constexpr (!A_F32) {
#pragma unroll
      for (int it = 0; it < AIT; ++it) {
        int e = tid * 8 + it * 2048, i = e >> 5, kx = e & 31;
        *(uint4*)&sA[nb][i * LDT + kx] = ra[it];
      }
    }
#pragma unroll
    for (int it = 0; it < BIT; ++it) {
      int e = tid * 8 + it * 2048, j = e >> 5, kx = e & 31;
      *(uint4*)&sB[nb][j * LDT + kx] = rb[it];
    }
#endif
  };

  // -------- pipelined main loop --------
  stage_issue(0, 0);
  stage_commit(0);
  __syncthreads();
  int cur = 0;
  for (int k0 = 0; k0 < K; k0 += BK) {
    const bool more = (k0 + BK) < K;
    if (more) stage_issue(k0 + BK, cur ^ 1);

    union FU { u32 u[8]; bf16x16 v; };
    FU fa[MI], fb[NI];
#pragma unroll
    for (int mi = 0; mi < MI; ++mi) {
      const u32* p = (const u32*)&sA[cur][(mW + mi * 16 + lrow) * LDT];
#pragma unroll
      for (int r = 0; r < 8; ++r)
        fa[mi].u[r] = p[((r & 4) << 1) + (half << 2) + (r & 3)];
    }
#pragma unroll
    for (int ni = 0; ni < NI; ++ni) {
      const u32* p = (const u32*)&sB[cur][(nW + ni * 16 + lrow) * LDT];
#pragma unroll
      for (int r = 0; r < 8; ++r)
        fb[ni].u[r] = p[((r & 4) << 1) + (half << 2) + (r & 3)];
    }
#pragma unroll
    for (int mi = 0; mi < MI; ++mi)
#pragma unroll
      for (int ni = 0; ni < NI; ++ni)
        acc[mi][ni] = __builtin_amdgcn_wmma_f32_16x16x32_bf16(
            false, fa[mi].v, false, fb[ni].v, (short)0, acc[mi][ni], false, false);

    if (more) {
      stage_commit(cur ^ 1);
      __syncthreads();
      cur ^= 1;
    }
  }

  // ---------------- epilogue ----------------
  u16*   outB = (u16*)Cg;
  float* outF = (float*)Cg;
  if constexpr (MODE == 6) {         // V -> (b,h,hd,t), 8 bf16 per uint4 store
#pragma unroll
    for (int mi = 0; mi < MI; ++mi)
#pragma unroll
      for (int ni = 0; ni < NI; ++ni) {
        int n = n0 + nW + ni * 16 + lrow;
        float bn = bias[n];
        int mb = m0 + mW + mi * 16 + half * 8;
        int b = mb >> 9, t = mb & (Tc - 1);
        int h = n >> 6, hd = n & (HDc - 1);
        uint4 st;
        st.x = pack2(acc[mi][ni][0] + bn, acc[mi][ni][1] + bn);
        st.y = pack2(acc[mi][ni][2] + bn, acc[mi][ni][3] + bn);
        st.z = pack2(acc[mi][ni][4] + bn, acc[mi][ni][5] + bn);
        st.w = pack2(acc[mi][ni][6] + bn, acc[mi][ni][7] + bn);
        *(uint4*)&outB[(((size_t)(b * Hc + h) * HDc) + hd) * Tc + t] = st;
      }
  } else {
#pragma unroll
    for (int mi = 0; mi < MI; ++mi)
#pragma unroll
      for (int ni = 0; ni < NI; ++ni)
#pragma unroll
        for (int r = 0; r < 8; ++r) {
          int m = m0 + mW + mi * 16 + r + half * 8;
          int n = n0 + nW + ni * 16 + lrow;
          float v = acc[mi][ni][r];
          if constexpr (MODE == 1) {           // Q/K -> (b,h,t,hd) bf16
            v += bias[n];
            int b = m >> 9, t = m & (Tc - 1);
            int h = n >> 6, hd = n & (HDc - 1);
            outB[(((size_t)(b * Hc + h) * Tc) + t) * HDc + hd] = f2bf(v);
          } else if constexpr (MODE == 2) {    // scores f32
            int h = z & (Hc - 1);
            v = v * scale + extra[((size_t)h * Tc + m) * Tc + n];
            outF[((size_t)z * Tc + m) * Tc + n] = v;
          } else if constexpr (MODE == 3) {    // attn@V -> (b,t,e) bf16
            int b = z / Hc, h = z % Hc;
            outB[((size_t)(b * Tc + m)) * Ec + h * HDc + n] = f2bf(v);
          } else if constexpr (MODE == 4) {    // +bias +residual f32
            v += bias[n] + extra[(size_t)m * ldc + n];
            outF[(size_t)m * ldc + n] = v;
          } else if constexpr (MODE == 5) {    // +bias, GELU, bf16
            v += bias[n];
            v = 0.5f * v * (1.0f + erff(v * 0.70710678118654752f));
            outB[(size_t)m * ldc + n] = f2bf(v);
          }
        }
  }
}

// ---------------- LayerNorm: fp32 row -> bf16 ----------------
__global__ __launch_bounds__(256)
void ln_kernel(const float* __restrict__ x, const float* __restrict__ g,
               const float* __restrict__ b, u16* __restrict__ out)
{
  __shared__ float r1[256], r2[256];
  const int row = blockIdx.x, tid = threadIdx.x;
  const float* xr = x + (size_t)row * Ec;
  float4 xv = *(const float4*)&xr[tid * 4];
  r1[tid] = xv.x + xv.y + xv.z + xv.w;
  r2[tid] = xv.x * xv.x + xv.y * xv.y + xv.z * xv.z + xv.w * xv.w;
  __syncthreads();
  for (int off = 128; off > 0; off >>= 1) {
    if (tid < off) { r1[tid] += r1[tid + off]; r2[tid] += r2[tid + off]; }
    __syncthreads();
  }
  float mean = r1[0] * (1.0f / Ec);
  float var  = r2[0] * (1.0f / Ec) - mean * mean;
  float inv  = rsqrtf(var + EPSc);
  int i = tid * 4;
  float vs[4] = {xv.x, xv.y, xv.z, xv.w};
#pragma unroll
  for (int u = 0; u < 4; ++u)
    out[(size_t)row * Ec + i + u] = f2bf((vs[u] - mean) * inv * g[i + u] + b[i + u]);
}

// ---------------- row softmax over T (in place, f32) ----------------
__global__ __launch_bounds__(256)
void softmax_kernel(float* __restrict__ s)
{
  __shared__ float red[256];
  const int tid = threadIdx.x;
  float* p = s + (size_t)blockIdx.x * Tc;
  float2 v = *(float2*)&p[tid * 2];
  red[tid] = fmaxf(v.x, v.y);
  __syncthreads();
  for (int off = 128; off > 0; off >>= 1) {
    if (tid < off) red[tid] = fmaxf(red[tid], red[tid + off]);
    __syncthreads();
  }
  float mx = red[0];
  __syncthreads();
  float e0 = expf(v.x - mx), e1 = expf(v.y - mx);
  red[tid] = e0 + e1;
  __syncthreads();
  for (int off = 128; off > 0; off >>= 1) {
    if (tid < off) red[tid] += red[tid + off];
    __syncthreads();
  }
  float rinv = 1.0f / red[0];
  float2 o; o.x = e0 * rinv; o.y = e1 * rinv;
  *(float2*)&p[tid * 2] = o;
}

// -------- interpolated relative-position bias -> [H][T][T] f32 --------
__global__ __launch_bounds__(256)
void tbias_kernel(const float* __restrict__ table, const float* __restrict__ offset,
                  float* __restrict__ out)
{
  int idx = blockIdx.x * 256 + threadIdx.x;            // over T*T
  int q = idx >> 9, k = idx & (Tc - 1);
  float off = tanhf(offset[0]) * MAXOFFc;
  float rel = (float)(q - k + (Tc - 1));
  float adj = fminf(fmaxf(rel + off, 0.0f), 2.0f * Tc - 2.0f);
  float lo = floorf(adj);
  float w  = adj - lo;
  int loi = (int)lo;
  int hii = (int)ceilf(adj);
#pragma unroll
  for (int h = 0; h < Hc; ++h) {
    float bv = table[loi * Hc + h] * (1.0f - w) + table[hii * Hc + h] * w;
    out[(size_t)h * Tc * Tc + idx] = bv;
  }
}

// ---- weight convert + transpose: f32 [K][N] -> bf16 [N][K], per layer z ----
__global__ __launch_bounds__(256)
void cvt_t_kernel(const float* __restrict__ in, u16* __restrict__ out, int Kd, int Nd)
{
  __shared__ u16 tile[32][33];
  const size_t mat = (size_t)Kd * Nd;
  const float* ip = in + (size_t)blockIdx.z * mat;
  u16* op = out + (size_t)blockIdx.z * mat;
  const int n0 = blockIdx.x * 32, k0 = blockIdx.y * 32;
  const int tx = threadIdx.x & 31, ty = threadIdx.x >> 5;
#pragma unroll
  for (int r = ty; r < 32; r += 8)
    tile[r][tx] = f2bf(ip[(size_t)(k0 + r) * Nd + n0 + tx]);
  __syncthreads();
#pragma unroll
  for (int r = ty; r < 32; r += 8)
    op[(size_t)(n0 + r) * Kd + k0 + tx] = tile[tx][r];
}

// ============================================================
extern "C" void kernel_launch(void* const* d_in, const int* in_sizes, int n_in,
                              void* d_out, int out_size, void* d_ws, size_t ws_size,
                              hipStream_t stream)
{
  (void)in_sizes; (void)n_in; (void)out_size; (void)ws_size;
  const float* x      = (const float*)d_in[0];
  const float* ln1_g  = (const float*)d_in[1];
  const float* ln1_b  = (const float*)d_in[2];
  const float* wq     = (const float*)d_in[3];
  const float* bq     = (const float*)d_in[4];
  const float* wk     = (const float*)d_in[5];
  const float* bk     = (const float*)d_in[6];
  const float* wv     = (const float*)d_in[7];
  const float* bv     = (const float*)d_in[8];
  const float* wo     = (const float*)d_in[9];
  const float* bo     = (const float*)d_in[10];
  const float* ln2_g  = (const float*)d_in[11];
  const float* ln2_b  = (const float*)d_in[12];
  const float* w1     = (const float*)d_in[13];
  const float* b1     = (const float*)d_in[14];
  const float* w2     = (const float*)d_in[15];
  const float* b2     = (const float*)d_in[16];
  const float* table  = (const float*)d_in[17];
  const float* offset = (const float*)d_in[18];

  // ---- workspace layout ----
  char* p = (char*)d_ws;
  auto alloc = [&](size_t bytes) { void* r = p; p += (bytes + 255) & ~(size_t)255; return r; };
  const size_t BT = (size_t)Bc * Tc;
  float* xbuf = (float*)alloc(BT * Ec * 4);
  u16*   hbf  = (u16*)  alloc(BT * Ec * 2);
  u16*   qb   = (u16*)  alloc(BT * Ec * 2);
  u16*   kb   = (u16*)  alloc(BT * Ec * 2);
  u16*   vb   = (u16*)  alloc(BT * Ec * 2);      // (b,h,hd,t)
  float* sc   = (float*)alloc((size_t)Bc * Hc * Tc * Tc * 4);
  u16*   ffb  = (u16*)  alloc(BT * FFc * 2);
  float* tb   = (float*)alloc((size_t)Hc * Tc * Tc * 4);
  u16*   wqb  = (u16*)  alloc((size_t)Lc * Ec * Ec * 2);    // transposed [N][K]
  u16*   wkb  = (u16*)  alloc((size_t)Lc * Ec * Ec * 2);
  u16*   wvb  = (u16*)  alloc((size_t)Lc * Ec * Ec * 2);
  u16*   wob  = (u16*)  alloc((size_t)Lc * Ec * Ec * 2);
  u16*   w1b  = (u16*)  alloc((size_t)Lc * Ec * FFc * 2);   // [FF][E]
  u16*   w2b  = (u16*)  alloc((size_t)Lc * FFc * Ec * 2);   // [E][FF]

  // ---- weights f32 -> bf16, transposed to [N][K] ----
  const dim3 blk(256);
  cvt_t_kernel<<<dim3(Ec / 32,  Ec / 32,  Lc), blk, 0, stream>>>(wq, wqb, Ec,  Ec);
  cvt_t_kernel<<<dim3(Ec / 32,  Ec / 32,  Lc), blk, 0, stream>>>(wk, wkb, Ec,  Ec);
  cvt_t_kernel<<<dim3(Ec / 32,  Ec / 32,  Lc), blk, 0, stream>>>(wv, wvb, Ec,  Ec);
  cvt_t_kernel<<<dim3(Ec / 32,  Ec / 32,  Lc), blk, 0, stream>>>(wo, wob, Ec,  Ec);
  cvt_t_kernel<<<dim3(FFc / 32, Ec / 32,  Lc), blk, 0, stream>>>(w1, w1b, Ec,  FFc);
  cvt_t_kernel<<<dim3(Ec / 32,  FFc / 32, Lc), blk, 0, stream>>>(w2, w2b, FFc, Ec);

  (void)hipMemcpyAsync(xbuf, x, BT * Ec * 4, hipMemcpyDeviceToDevice, stream);

  const dim3 gEE(Ec / 128, (int)(BT / 128), 1);        // (8, 64)
  const dim3 gEF(FFc / 128, (int)(BT / 128), 1);       // (32, 64)
  const dim3 gScore(Tc / 128, Tc / 128, Bc * Hc);      // (4, 4, 256)
  const dim3 gAV(1, Tc / 128, Bc * Hc);                // (1, 4, 256)

  for (int l = 0; l < Lc; ++l) {
    const size_t oEE = (size_t)l * Ec * Ec, oEF = (size_t)l * Ec * FFc;
    // LN1
    ln_kernel<<<(int)BT, blk, 0, stream>>>(xbuf, ln1_g + l * Ec, ln1_b + l * Ec, hbf);
    // Q, K projections -> (b,h,t,hd); V -> (b,h,hd,t)
    gemm_wmma<128,128,2,4,1,false><<<gEE, blk, 0, stream>>>(
        hbf, wqb + oEE, bq + l * Ec, nullptr, qb, Ec, Ec, Ec, Ec, 0, 0, 1.0f);
    gemm_wmma<128,128,2,4,1,false><<<gEE, blk, 0, stream>>>(
        hbf, wkb + oEE, bk + l * Ec, nullptr, kb, Ec, Ec, Ec, Ec, 0, 0, 1.0f);
    gemm_wmma<128,128,2,4,6,false><<<gEE, blk, 0, stream>>>(
        hbf, wvb + oEE, bv + l * Ec, nullptr, vb, Ec, Ec, Ec, Ec, 0, 0, 1.0f);
    // temporal bias for this layer
    tbias_kernel<<<Tc * Tc / 256, blk, 0, stream>>>(
        table + (size_t)l * (2 * Tc - 1) * Hc, offset + l, tb);
    // scores = Q K^T * scale + bias   (per (b,h) slice; K is [N=t][K=hd])
    gemm_wmma<128,128,2,4,2,false><<<gScore, blk, 0, stream>>>(
        qb, kb, nullptr, tb, sc, HDc, HDc, HDc, Tc,
        (long long)Tc * HDc, (long long)Tc * HDc, SCALEc);
    // softmax over keys
    softmax_kernel<<<Bc * Hc * Tc, blk, 0, stream>>>(sc);
    // out = P V  (A = f32 probs -> bf16 in staging; V^T is [N=hd][K=t])
    gemm_wmma<128,64,4,2,3,true><<<gAV, blk, 0, stream>>>(
        sc, vb, nullptr, nullptr, hbf, Tc, Tc, Tc, HDc,
        (long long)Tc * Tc, (long long)HDc * Tc, 1.0f);
    // x += out @ Wo + bo
    gemm_wmma<128,128,2,4,4,false><<<gEE, blk, 0, stream>>>(
        hbf, wob + oEE, bo + l * Ec, xbuf, xbuf, Ec, Ec, Ec, Ec, 0, 0, 1.0f);
    // LN2
    ln_kernel<<<(int)BT, blk, 0, stream>>>(xbuf, ln2_g + l * Ec, ln2_b + l * Ec, hbf);
    // ff = gelu(h @ W1 + b1)
    gemm_wmma<128,128,2,4,5,false><<<gEF, blk, 0, stream>>>(
        hbf, w1b + oEF, b1 + l * FFc, nullptr, ffb, Ec, Ec, FFc, FFc, 0, 0, 1.0f);
    // x += ff @ W2 + b2
    gemm_wmma<128,128,2,4,4,false><<<gEE, blk, 0, stream>>>(
        ffb, w2b + oEF, b2 + l * Ec, xbuf, xbuf, FFc, FFc, Ec, Ec, 0, 0, 1.0f);
  }

  (void)hipMemcpyAsync(d_out, xbuf, BT * Ec * 4, hipMemcpyDeviceToDevice, stream);
}
